// binary_disordered_RNNwavefunction_weight_sharing_90881507983705
// MI455X (gfx1250) — matrix operand order repaired and checked by reading, not compiled, not run on hardware
//
#include <hip/hip_runtime.h>

#define Bv 8192
#define Nv 1024
#define Hv 64

typedef __attribute__((ext_vector_type(16))) __bf16 v16bf;
typedef __attribute__((ext_vector_type(8)))  float  v8f;

union FragU { uint4 u[2]; v16bf v; };

// ---------------- LDS layout (bytes) ----------------
#define OFF_WHH0   0        // 192x64 bf16
#define OFF_WIH1   24576    // 192x64 bf16
#define OFF_WHH1   49152    // 192x64 bf16
#define OFF_WD1    73728    // 64x64  bf16
#define OFF_WD2P   81920    // 16x64  bf16 (rows 2..15 zero-padded)
#define OFF_WIH0   83968    // 192x2  f32
#define OFF_BRZ0   85504    // 128 f32: b_ih0[j]+b_hh0[j], j<128 (r,z gates)
#define OFF_BNI0   86016    // 64 f32: b_ih0[128+j]
#define OFF_BNH0   86272    // 64 f32: b_hh0[128+j]
#define OFF_BRZ1   86528    // 128 f32
#define OFF_BNI1   87040    // 64 f32
#define OFF_BNH1   87296    // 64 f32
#define OFF_BD1    87552    // 64 f32
#define OFF_BD2    87808    // 2 f32 (pad 16)
#define OFF_STAGE  87824
// per-wave staging region:
#define WS_H0A     0        // 16x64 bf16 tile (2048B)  (H0B = H0A ^ 2048)
#define WS_H1A     4096
#define WS_HID     8192
#define WS_X0      10240    // 16 f32
#define WS_X1      10304    // 16 f32
#define WS_LOGITS  10368    // 32 f32
#define WSTRIDE    10496
#define SMEM_TOTAL (OFF_STAGE + 4*WSTRIDE)   // 129808 bytes

// ---------------- helpers ----------------
static __device__ __forceinline__ unsigned short f2bf(float f){
  return (unsigned short)(__float_as_uint(f) >> 16);
}
static __device__ __forceinline__ float sigm(float x){ return 1.0f/(1.0f+__expf(-x)); }
static __device__ __forceinline__ float tanh_f(float x){ return 2.0f*sigm(2.0f*x) - 1.0f; }

static __device__ __forceinline__ float rng01(unsigned row, unsigned step){
  unsigned x = row * 2654435761u + step * 0x9E3779B9u + 0x85EBCA6Bu;
  x ^= x >> 16; x *= 0x7FEB352Du;
  x ^= x >> 15; x *= 0x846CA68Bu;
  x ^= x >> 16;
  return (float)(x >> 8) * (1.0f/16777216.0f);
}

static __device__ __forceinline__ v8f zero8(){
  v8f z = {0.f,0.f,0.f,0.f,0.f,0.f,0.f,0.f};
  return z;
}
static __device__ __forceinline__ v8f wmma_bf16(v16bf a, v16bf b, v8f c){
  return __builtin_amdgcn_wmma_f32_16x16x32_bf16(false, a, false, b, (short)0, c, false, false);
}

// A fragment: 16x32 bf16 tile from row-major 16x64 bf16 LDS array (128B row stride).
static __device__ __forceinline__ v16bf fragA(const char* base, int kt, int lane){
  int m = lane & 15, g = lane >> 4;
  const char* p = base + m*128 + kt*64 + g*16;
  FragU f;
  f.u[0] = *(const uint4*)p;
  f.u[1] = *(const uint4*)(p + 32);
  return f.v;
}

// B fragment: 32x16 bf16 tile. B[k][n] = W[j0+n][32*kt + k], W row-major 64-wide.
static __device__ __forceinline__ v16bf fragB(const char* wbase, int j0, int kt, int lane){
  int n = lane & 15, g = lane >> 4;
  const char* p = wbase + (j0 + n)*128 + kt*64 + g*16;
  FragU f;
  f.u[0] = *(const uint4*)p;
  f.u[1] = *(const uint4*)(p + 32);
  return f.v;
}

// ---------------- GRU cell 1: gi via matmul; r/z accumulate gi+gh in ONE acc ----------------
static __device__ __forceinline__ void gru_mm(const char* aIn, const char* hOld, char* hNew,
                                              const char* wIH, const char* wHH,
                                              const float* cRZr, const float* cRZz,
                                              const float* cNi,  const float* cNh,
                                              float* hPrev, int lane){
  v16bf a0 = fragA(aIn, 0, lane), a1 = fragA(aIn, 1, lane);
  v16bf hA0 = fragA(hOld, 0, lane), hA1 = fragA(hOld, 1, lane);
  int n = lane & 15, g = lane >> 4;
  #pragma unroll
  for (int t = 0; t < 4; ++t){
    int j0r = t*16, j0z = 64 + t*16, j0n = 128 + t*16;
    v8f aR = zero8();
    aR = wmma_bf16(a0,  fragB(wIH, j0r, 0, lane), aR);
    aR = wmma_bf16(a1,  fragB(wIH, j0r, 1, lane), aR);
    aR = wmma_bf16(hA0, fragB(wHH, j0r, 0, lane), aR);
    aR = wmma_bf16(hA1, fragB(wHH, j0r, 1, lane), aR);
    v8f aZ = zero8();
    aZ = wmma_bf16(a0,  fragB(wIH, j0z, 0, lane), aZ);
    aZ = wmma_bf16(a1,  fragB(wIH, j0z, 1, lane), aZ);
    aZ = wmma_bf16(hA0, fragB(wHH, j0z, 0, lane), aZ);
    aZ = wmma_bf16(hA1, fragB(wHH, j0z, 1, lane), aZ);
    v8f giN = zero8();
    giN = wmma_bf16(a0, fragB(wIH, j0n, 0, lane), giN);
    giN = wmma_bf16(a1, fragB(wIH, j0n, 1, lane), giN);
    v8f ghN = zero8();
    ghN = wmma_bf16(hA0, fragB(wHH, j0n, 0, lane), ghN);
    ghN = wmma_bf16(hA1, fragB(wHH, j0n, 1, lane), ghN);

    #pragma unroll
    for (int r = 0; r < 8; ++r){
      int m = r + 8*g;
      float rr = sigm(aR[r] + cRZr[t]);
      float zz = sigm(aZ[r] + cRZz[t]);
      float nn = tanh_f(giN[r] + cNi[t] + rr*(ghN[r] + cNh[t]));
      float ho = hPrev[t*8 + r];
      float hv = fmaf(zz, ho - nn, nn);
      hPrev[t*8 + r] = hv;
      *(unsigned short*)(hNew + m*128 + (t*16+n)*2) = f2bf(hv);
    }
  }
}

// ---------------- GRU cell 0: gi is a gather (x is one-hot / zero, D=2) ----------------
static __device__ __forceinline__ void gru0(const float* x0, const float* x1,
                                            const char* hOld, char* hNew,
                                            const char* wHH, const float* wIH0,
                                            const float* cRZr, const float* cRZz,
                                            const float* cNi,  const float* cNh,
                                            float* hPrev, int lane){
  v16bf hA0 = fragA(hOld, 0, lane), hA1 = fragA(hOld, 1, lane);
  int n = lane & 15, g = lane >> 4;
  float xa0[8], xa1[8];
  #pragma unroll
  for (int r = 0; r < 8; ++r){ xa0[r] = x0[r + 8*g]; xa1[r] = x1[r + 8*g]; }
  #pragma unroll
  for (int t = 0; t < 4; ++t){
    int j0r = t*16, j0z = 64 + t*16, j0n = 128 + t*16;
    v8f aR = zero8();
    aR = wmma_bf16(hA0, fragB(wHH, j0r, 0, lane), aR);
    aR = wmma_bf16(hA1, fragB(wHH, j0r, 1, lane), aR);
    v8f aZ = zero8();
    aZ = wmma_bf16(hA0, fragB(wHH, j0z, 0, lane), aZ);
    aZ = wmma_bf16(hA1, fragB(wHH, j0z, 1, lane), aZ);
    v8f ghN = zero8();
    ghN = wmma_bf16(hA0, fragB(wHH, j0n, 0, lane), ghN);
    ghN = wmma_bf16(hA1, fragB(wHH, j0n, 1, lane), ghN);

    int j = t*16 + n;
    float wr0 = wIH0[j*2],        wr1 = wIH0[j*2 + 1];
    float wz0 = wIH0[(64+j)*2],   wz1 = wIH0[(64+j)*2 + 1];
    float wn0 = wIH0[(128+j)*2],  wn1 = wIH0[(128+j)*2 + 1];
    #pragma unroll
    for (int r = 0; r < 8; ++r){
      int m = r + 8*g;
      float gi_r = fmaf(xa0[r], wr0, xa1[r]*wr1);
      float gi_z = fmaf(xa0[r], wz0, xa1[r]*wz1);
      float gi_n = fmaf(xa0[r], wn0, xa1[r]*wn1);
      float rr = sigm(aR[r] + gi_r + cRZr[t]);
      float zz = sigm(aZ[r] + gi_z + cRZz[t]);
      float nn = tanh_f(gi_n + cNi[t] + rr*(ghN[r] + cNh[t]));
      float ho = hPrev[t*8 + r];
      float hv = fmaf(zz, ho - nn, nn);
      hPrev[t*8 + r] = hv;
      *(unsigned short*)(hNew + m*128 + (t*16+n)*2) = f2bf(hv);
    }
  }
}

static __device__ __forceinline__ void dense_relu(const char* a, char* out, const char* W,
                                                  const float* cB, int lane){
  v16bf a0 = fragA(a, 0, lane), a1 = fragA(a, 1, lane);
  int n = lane & 15, g = lane >> 4;
  #pragma unroll
  for (int t = 0; t < 4; ++t){
    v8f acc = zero8();
    acc = wmma_bf16(a0, fragB(W, t*16, 0, lane), acc);
    acc = wmma_bf16(a1, fragB(W, t*16, 1, lane), acc);
    #pragma unroll
    for (int r = 0; r < 8; ++r){
      int m = r + 8*g;
      float v = fmaxf(acc[r] + cB[t], 0.0f);
      *(unsigned short*)(out + m*128 + (t*16+n)*2) = f2bf(v);
    }
  }
}

// ---------------- kernel ----------------
__global__ void __launch_bounds__(128, 1)
rnn_sampler_kernel(const float* __restrict__ inp,
                   const float* __restrict__ g_wih0, const float* __restrict__ g_whh0,
                   const float* __restrict__ g_bih0, const float* __restrict__ g_bhh0,
                   const float* __restrict__ g_wih1, const float* __restrict__ g_whh1,
                   const float* __restrict__ g_bih1, const float* __restrict__ g_bhh1,
                   const float* __restrict__ g_dw1,  const float* __restrict__ g_db1,
                   const float* __restrict__ g_dw2,  const float* __restrict__ g_db2,
                   float* __restrict__ out){
  extern __shared__ char smem[];
  const int tid = threadIdx.x, NT = blockDim.x;
  const int lane = tid & 31, wave = tid >> 5;
  const int rowBase0 = blockIdx.x * 64;

  // ---- stage weights: f32 -> bf16 into LDS ----
  unsigned short* wHH0 = (unsigned short*)(smem + OFF_WHH0);
  unsigned short* wIH1 = (unsigned short*)(smem + OFF_WIH1);
  unsigned short* wHH1 = (unsigned short*)(smem + OFF_WHH1);
  unsigned short* wD1  = (unsigned short*)(smem + OFF_WD1);
  unsigned short* wD2p = (unsigned short*)(smem + OFF_WD2P);
  float* wIH0 = (float*)(smem + OFF_WIH0);
  float* bRZ0 = (float*)(smem + OFF_BRZ0);
  float* bNi0 = (float*)(smem + OFF_BNI0);
  float* bNh0 = (float*)(smem + OFF_BNH0);
  float* bRZ1 = (float*)(smem + OFF_BRZ1);
  float* bNi1 = (float*)(smem + OFF_BNI1);
  float* bNh1 = (float*)(smem + OFF_BNH1);
  float* bD1  = (float*)(smem + OFF_BD1);
  float* bD2  = (float*)(smem + OFF_BD2);

  for (int i = tid; i < 192*64; i += NT){
    wHH0[i] = f2bf(g_whh0[i]);
    wIH1[i] = f2bf(g_wih1[i]);
    wHH1[i] = f2bf(g_whh1[i]);
  }
  for (int i = tid; i < 64*64; i += NT) wD1[i] = f2bf(g_dw1[i]);
  for (int i = tid; i < 16*64; i += NT){
    int j = i >> 6, k = i & 63;
    wD2p[i] = (j < 2) ? f2bf(g_dw2[j*64 + k]) : (unsigned short)0;
  }
  for (int i = tid; i < 192*2; i += NT) wIH0[i] = g_wih0[i];
  for (int i = tid; i < 128; i += NT){
    bRZ0[i] = g_bih0[i] + g_bhh0[i];
    bRZ1[i] = g_bih1[i] + g_bhh1[i];
  }
  for (int i = tid; i < 64; i += NT){
    bNi0[i] = g_bih0[128 + i]; bNh0[i] = g_bhh0[128 + i];
    bNi1[i] = g_bih1[128 + i]; bNh1[i] = g_bhh1[128 + i];
    bD1[i]  = g_db1[i];
  }
  if (tid < 2) bD2[tid] = g_db2[tid];

  // ---- zero h staging (h0/h1 double buffers + hid) ----
  char* stage = smem + OFF_STAGE;
  for (int i = tid; i < 4*2560; i += NT){
    int w = i / 2560, off = i % 2560;
    ((unsigned*)(stage + w*WSTRIDE))[off] = 0u;
  }
  // ---- initial x from inputs ----
  for (int i = tid; i < 64; i += NT){
    int w = i >> 4, sl = i & 15;
    float* px0 = (float*)(stage + w*WSTRIDE + WS_X0);
    float* px1 = (float*)(stage + w*WSTRIDE + WS_X1);
    px0[sl] = inp[(rowBase0 + i)*2 + 0];
    px1[sl] = inp[(rowBase0 + i)*2 + 1];
  }
  __syncthreads();

  // ---- per-wave persistent state ----
  char* ws = stage + wave*WSTRIDE;
  char* hid = ws + WS_HID;
  float* x0 = (float*)(ws + WS_X0);
  float* x1 = (float*)(ws + WS_X1);
  float* logits = (float*)(ws + WS_LOGITS);
  const int rowBase = rowBase0 + wave*16;
  const int n = lane & 15, g = lane >> 4;

  // hoist per-lane loop-invariant gate constants (depend only on lane/t)
  float cRZr0[4], cRZz0[4], cNi0[4], cNh0[4];
  float cRZr1[4], cRZz1[4], cNi1[4], cNh1[4];
  float cBD1[4];
  #pragma unroll
  for (int t = 0; t < 4; ++t){
    int j = t*16 + n;
    cRZr0[t] = bRZ0[j];      cRZz0[t] = bRZ0[64 + j];
    cNi0[t]  = bNi0[j];      cNh0[t]  = bNh0[j];
    cRZr1[t] = bRZ1[j];      cRZz1[t] = bRZ1[64 + j];
    cNi1[t]  = bNi1[j];      cNh1[t]  = bNh1[j];
    cBD1[t]  = bD1[j];
  }
  float cBD2 = (n < 2) ? bD2[n] : 0.0f;

  // previous-step h values live in registers
  float h0p[32], h1p[32];
  #pragma unroll
  for (int i = 0; i < 32; ++i){ h0p[i] = 0.0f; h1p[i] = 0.0f; }

  float lp = 0.0f;

  for (int step = 0; step < Nv; ++step){
    int par = (step & 1) << 11;                 // 0 or 2048
    char* h0o = ws + (WS_H0A + par);
    char* h0n = ws + (WS_H0A + (par ^ 2048));
    char* h1o = ws + (WS_H1A + par);
    char* h1n = ws + (WS_H1A + (par ^ 2048));

    gru0(x0, x1, h0o, h0n, (const char*)wHH0, wIH0,
         cRZr0, cRZz0, cNi0, cNh0, h0p, lane);
    __builtin_amdgcn_wave_barrier();
    gru_mm(h0n, h1o, h1n, (const char*)wIH1, (const char*)wHH1,
           cRZr1, cRZz1, cNi1, cNh1, h1p, lane);
    __builtin_amdgcn_wave_barrier();
    dense_relu(h1n, hid, (const char*)wD1, cBD1, lane);
    __builtin_amdgcn_wave_barrier();

    // output head: (16x64) x (64x2) via one WMMA tile (cols >=2 zero-padded)
    {
      v16bf a0 = fragA(hid, 0, lane), a1 = fragA(hid, 1, lane);
      v8f acc = zero8();
      acc = wmma_bf16(a0, fragB((const char*)wD2p, 0, 0, lane), acc);
      acc = wmma_bf16(a1, fragB((const char*)wD2p, 0, 1, lane), acc);
      if (n < 2){
        #pragma unroll
        for (int r = 0; r < 8; ++r) logits[(r + 8*g)*2 + n] = acc[r] + cBD2;
      }
    }
    __builtin_amdgcn_wave_barrier();

    // sample: lanes 0..15 each own one batch row of this tile
    if (lane < 16){
      float l0 = logits[2*lane], l1 = logits[2*lane + 1];
      float p1 = sigm(l1 - l0) + 1e-10f;        // softmax over 2 = sigmoid
      float p0 = (1.0f - (p1 - 1e-10f)) + 1e-10f;
      unsigned grow = (unsigned)(rowBase + lane);
      float u = rng01(grow, (unsigned)step);
      int s = (u * (p0 + p1) < p0) ? 0 : 1;
      float ps = s ? p1 : p0;
      lp += __logf(ps + 1e-10f);
      x0[lane] = s ? 0.0f : 1.0f;
      x1[lane] = s ? 1.0f : 0.0f;
      out[(size_t)grow * Nv + step] = (float)s;
    }
    __builtin_amdgcn_wave_barrier();
  }

  if (lane < 16){
    out[(size_t)Bv * Nv + (rowBase + lane)] = lp;
  }
}

// ---------------- launcher ----------------
extern "C" void kernel_launch(void* const* d_in, const int* in_sizes, int n_in,
                              void* d_out, int out_size, void* d_ws, size_t ws_size,
                              hipStream_t stream) {
  (void)in_sizes; (void)n_in; (void)out_size; (void)d_ws; (void)ws_size;
  const float* inp   = (const float*)d_in[0];
  const float* wih0  = (const float*)d_in[1];
  const float* whh0  = (const float*)d_in[2];
  const float* bih0  = (const float*)d_in[3];
  const float* bhh0  = (const float*)d_in[4];
  const float* wih1  = (const float*)d_in[5];
  const float* whh1  = (const float*)d_in[6];
  const float* bih1  = (const float*)d_in[7];
  const float* bhh1  = (const float*)d_in[8];
  const float* dw1   = (const float*)d_in[9];
  const float* db1   = (const float*)d_in[10];
  const float* dw2   = (const float*)d_in[11];
  const float* db2   = (const float*)d_in[12];
  float* out = (float*)d_out;

  size_t shmem = SMEM_TOTAL;
  hipFuncSetAttribute((const void*)rnn_sampler_kernel,
                      hipFuncAttributeMaxDynamicSharedMemorySize, (int)shmem);

  dim3 grid(Bv / 64), block(128);
  rnn_sampler_kernel<<<grid, block, shmem, stream>>>(
      inp, wih0, whh0, bih0, bhh0, wih1, whh1, bih1, bhh1, dw1, db1, dw2, db2, out);
}
